// SSEGCNBertClassifier_7284264534267
// MI455X (gfx1250) — matrix operand order repaired and checked by reading, not compiled, not run on hardware
//
#include <hip/hip_runtime.h>
#include <hip/hip_bf16.h>

// ---------------------------------------------------------------------------
// Types for CDNA5 WMMA (wave32, 16x16x32 f16 -> f32)
// ---------------------------------------------------------------------------
typedef __attribute__((ext_vector_type(16))) _Float16 v16h;
typedef __attribute__((ext_vector_type(8)))  _Float16 v8h;
typedef __attribute__((ext_vector_type(8)))  float    v8f;

#define NP 112   // padded N (100 -> 7 tiles of 16)
#define KP_S 128 // padded K for 100-wide activations

// A-matrix 16x32 f16 fragment: lane = (half<<4)|row ; element j:
//   j<8  -> K = half*8 + j ; j>=8 -> K = 16 + half*8 + (j-8)
__device__ inline v16h load_fragA(const _Float16* rowPtr, int half) {
  v8h lo = *(const v8h*)(rowPtr + half * 8);
  v8h hi = *(const v8h*)(rowPtr + 16 + half * 8);
  v16h f;
#pragma unroll
  for (int i = 0; i < 8; ++i) { f[i] = lo[i]; f[i + 8] = hi[i]; }
  return f;
}

// B-matrix 32x16 f16 fragment: lane&15 = N column, element j:
//   j<8  -> K = half*8 + j ; j>=8 -> K = 16 + half*8 + (j-8)
// colPtr points at column n's contiguous K-run (we store B transposed: [n][k]).
__device__ inline v16h load_fragB(const _Float16* colPtr, int half) {
  v8h lo = *(const v8h*)(colPtr + half * 8);
  v8h hi = *(const v8h*)(colPtr + 16 + half * 8);
  v16h f;
#pragma unroll
  for (int i = 0; i < 8; ++i) { f[i] = lo[i]; f[i + 8] = hi[i]; }
  return f;
}

// ---------------------------------------------------------------------------
// Utility kernels
// ---------------------------------------------------------------------------
__global__ void zero_h_k(_Float16* p, size_t n) {
  size_t i = (size_t)blockIdx.x * blockDim.x + threadIdx.x;
  if (i < n) p[i] = (_Float16)0.0f;
}

// Pack W (Kreal x Nreal, row-major f32) -> Wt (112 x KP, f16, transposed, zero pad)
__global__ void packWT_k(const float* __restrict__ W, _Float16* __restrict__ Wt,
                         int Kreal, int Nreal, int KP) {
  int idx = blockIdx.x * blockDim.x + threadIdx.x;
  if (idx >= NP * KP) return;
  int n = idx / KP, k = idx % KP;
  float v = (k < Kreal && n < Nreal) ? W[(size_t)k * Nreal + n] : 0.0f;
  Wt[idx] = (_Float16)v;
}

__global__ void packBias_k(const float* __restrict__ b, float* __restrict__ out, int Nreal) {
  int i = threadIdx.x;
  if (i < NP) out[i] = (i < Nreal) ? b[i] : 0.0f;
}

// ---------------------------------------------------------------------------
// LayerNorm (ddof=1) -> f16, one wave per row of 768
// ---------------------------------------------------------------------------
__global__ void __launch_bounds__(256) layernorm_k(
    const float* __restrict__ x, const float* __restrict__ ln_a,
    const float* __restrict__ ln_b, _Float16* __restrict__ xh) {
  int wave = __builtin_amdgcn_readfirstlane(threadIdx.x >> 5);
  int lane = threadIdx.x & 31;
  int row = blockIdx.x * 8 + wave;
  const float* xr = x + (size_t)row * 768;
  float v[24], s = 0.f, ss = 0.f;
#pragma unroll
  for (int k = 0; k < 24; ++k) { v[k] = xr[lane + 32 * k]; s += v[k]; ss += v[k] * v[k]; }
  for (int off = 16; off > 0; off >>= 1) { s += __shfl_xor(s, off, 32); ss += __shfl_xor(ss, off, 32); }
  float mean = s * (1.0f / 768.0f);
  float var  = (ss - 768.0f * mean * mean) * (1.0f / 767.0f);
  float inv  = 1.0f / (sqrtf(fmaxf(var, 0.0f)) + 1e-6f);
#pragma unroll
  for (int k = 0; k < 24; ++k) {
    int c = lane + 32 * k;
    xh[(size_t)row * 768 + c] = (_Float16)(ln_a[c] * (v[k] - mean) * inv + ln_b[c]);
  }
}

// ---------------------------------------------------------------------------
// Dense GEMM: C[M x 112] = A[M x KP] @ Wt^T + bias, optional relu.
// Block = 224 threads = 7 waves; wave w owns N-tile w (structurally uniform,
// no EXEC masking around WMMA). 16 rows of C per block.
// Outputs: f32 (stride 112), f16 (stride 128), per-batch transposed f16
// ([b][112][256]), per-head padded f16 q/k ([(b*5+h)*256+l][32]).
// ---------------------------------------------------------------------------
__global__ void __launch_bounds__(224) dense_gemm_k(
    const _Float16* __restrict__ A, int KP,
    const _Float16* __restrict__ Wt, const float* __restrict__ bias, int doRelu,
    float* __restrict__ outF, _Float16* __restrict__ outH,
    _Float16* __restrict__ outT, _Float16* __restrict__ outPH) {
  const int lane = threadIdx.x & 31;
  const int t    = __builtin_amdgcn_readfirstlane(threadIdx.x >> 5);  // N-tile 0..6
  const int half = lane >> 4;
  const int rl   = lane & 15;
  const int m0   = blockIdx.x * 16;

  v8f acc = {};
  const _Float16* aRow = A + (size_t)(m0 + rl) * KP;
  const _Float16* bRow = Wt + (size_t)(t * 16 + rl) * KP;
  for (int k0 = 0; k0 < KP; k0 += 32) {
    v16h af = load_fragA(aRow + k0, half);
    v16h bf = load_fragB(bRow + k0, half);
    acc = __builtin_amdgcn_wmma_f32_16x16x32_f16(false, af, false, bf,
                                                 (short)0, acc, false, false);
  }
  const int n = t * 16 + rl;
  const float bv = bias[n];
#pragma unroll
  for (int r = 0; r < 8; ++r) {
    int m = m0 + r + 8 * half;
    float v = acc[r] + bv;
    if (doRelu) v = fmaxf(v, 0.0f);
    if (outF) outF[(size_t)m * NP + n] = v;
    if (outH) outH[(size_t)m * KP_S + n] = (_Float16)v;
    if (outT) {
      int b = m >> 8, l = m & 255;
      outT[((size_t)b * NP + n) * 256 + l] = (_Float16)v;
    }
    if (outPH && n < 100) {
      int b = m >> 8, l = m & 255;
      int h = n / 20, d = n % 20;
      outPH[(((size_t)(b * 5 + h)) * 256 + l) * 32 + d] = (_Float16)v;
    }
  }
}

// ---------------------------------------------------------------------------
// aspect = (g * aspect_mask).sum(L)/wn ; asp = aspect @ dense_w + dense_b
// ---------------------------------------------------------------------------
__global__ void __launch_bounds__(128) aspect_k(
    const float* __restrict__ gF, const float* __restrict__ amask,
    const float* __restrict__ dense_w, const float* __restrict__ dense_b,
    float* __restrict__ aspOut) {
  __shared__ float aspect[100];
  __shared__ float red[128];
  int b = blockIdx.x, t = threadIdx.x;
  float wn = 0.f;
  for (int l = t; l < 256; l += 128) wn += amask[b * 256 + l];
  red[t] = wn; __syncthreads();
  for (int off = 64; off > 0; off >>= 1) {
    if (t < off) red[t] += red[t + off];
    __syncthreads();
  }
  wn = red[0];
  if (t < 100) {
    float s = 0.f;
    for (int l = 0; l < 256; ++l)
      s += gF[((size_t)b * 256 + l) * NP + t] * amask[b * 256 + l];
    aspect[t] = s / wn;
  }
  __syncthreads();
  if (t < 20) {
    float s = 0.f;
    for (int c = 0; c < 100; ++c) s += aspect[c] * dense_w[c * 20 + t];
    aspOut[b * 20 + t] = s + dense_b[t];
  }
}

// ---------------------------------------------------------------------------
// Attention: per (b,h) 16-row strip. QK^T via one WMMA per 16x16 tile (K=32,
// DK padded with zeros), + tanh(asp.k + bias) + mask + short_mask, softmax.
// 128 threads = 4 waves; each wave does 4 N-tiles (uniform trip count).
// ---------------------------------------------------------------------------
__global__ void __launch_bounds__(128) attention_k(
    const _Float16* __restrict__ qh, const _Float16* __restrict__ kh,
    const float* __restrict__ asp, const float* __restrict__ bias_m,
    const int* __restrict__ src_mask, const float* __restrict__ short_mask,
    float* __restrict__ adj) {
  __shared__ float sc[16][256];
  __shared__ float aspsc[256];
  const int tid  = threadIdx.x;
  const int lane = tid & 31;
  const int wave = __builtin_amdgcn_readfirstlane(tid >> 5);
  const int half = lane >> 4, rl = lane & 15;
  const int bh = blockIdx.x >> 4;         // b*5+h
  const int i0 = (blockIdx.x & 15) * 16;
  const int b  = bh / 5;

  for (int j = tid; j < 256; j += 128) {
    const _Float16* kr = kh + ((size_t)bh * 256 + j) * 32;
    float s = 0.f;
    for (int d = 0; d < 20; ++d) s += asp[b * 20 + d] * (float)kr[d];
    aspsc[j] = tanhf(s + bias_m[0]);
  }

  v16h af = load_fragA(qh + ((size_t)bh * 256 + i0 + rl) * 32, half);
  const float sqk = 0.22360679774997896f;  // 1/sqrt(20)
#pragma unroll
  for (int s = 0; s < 4; ++s) {
    int t = wave + 4 * s;
    v16h bf = load_fragB(kh + ((size_t)bh * 256 + t * 16 + rl) * 32, half);
    v8f acc = {};
    acc = __builtin_amdgcn_wmma_f32_16x16x32_f16(false, af, false, bf,
                                                 (short)0, acc, false, false);
#pragma unroll
    for (int r = 0; r < 8; ++r) sc[r + 8 * half][t * 16 + rl] = acc[r] * sqk;
  }
  __syncthreads();

  for (int rr = 0; rr < 4; ++rr) {
    int lr = wave + 4 * rr;
    int i  = i0 + lr;
    float vals[8], mx = -3.0e38f;
#pragma unroll
    for (int s = 0; s < 8; ++s) {
      int j = lane + 32 * s;
      float v = sc[lr][j] + aspsc[j];
      if (src_mask[b * 256 + j] == 0) v = -1.0e9f;
      v += short_mask[((size_t)b * 256 + i) * 256 + j];
      vals[s] = v; mx = fmaxf(mx, v);
    }
    for (int off = 16; off > 0; off >>= 1) mx = fmaxf(mx, __shfl_xor(mx, off, 32));
    float sum = 0.f;
#pragma unroll
    for (int s = 0; s < 8; ++s) { vals[s] = __expf(vals[s] - mx); sum += vals[s]; }
    for (int off = 16; off > 0; off >>= 1) sum += __shfl_xor(sum, off, 32);
    float inv = 1.0f / sum;
#pragma unroll
    for (int s = 0; s < 8; ++s)
      adj[((size_t)bh * 256 + i) * 256 + lane + 32 * s] = vals[s] * inv;
  }
}

// ---------------------------------------------------------------------------
// Head reductions: adjS = sum_h adj ; adjW = sum_h adj * rowsum(Wa)[h]
// (rowsum(Wa)[h] = sum_k Wx_w[h*5+k], Wa = Wx_w[:5])
// ---------------------------------------------------------------------------
__global__ void head_reduce_k(const float* __restrict__ adj,
                              const float* __restrict__ Wx_w,
                              _Float16* __restrict__ adjSh,
                              _Float16* __restrict__ adjWh) {
  size_t idx = (size_t)blockIdx.x * blockDim.x + threadIdx.x;
  if (idx >= (size_t)32 * 65536) return;
  size_t b = idx >> 16, ij = idx & 65535;
  const float* p = adj + b * 5 * 65536 + ij;
  float s = 0.f, w = 0.f;
#pragma unroll
  for (int h = 0; h < 5; ++h) {
    float wa = 0.f;
#pragma unroll
    for (int k = 0; k < 5; ++k) wa += Wx_w[h * 5 + k];
    float a = p[(size_t)h * 65536];
    s += a; w += a * wa;
  }
  adjSh[idx] = (_Float16)s;
  adjWh[idx] = (_Float16)w;
}

// ---------------------------------------------------------------------------
// Batched GEMM: Ax[b][256][<=112] = scale * adj[b][256][256] @ goT[b]^T
// 224 threads = 7 waves, one N-tile per wave (uniform).
// ---------------------------------------------------------------------------
__global__ void __launch_bounds__(224) bgemm_k(
    const _Float16* __restrict__ Aadj, const _Float16* __restrict__ Bt,
    float scale, _Float16* __restrict__ outH) {
  const int lane = threadIdx.x & 31;
  const int t    = __builtin_amdgcn_readfirstlane(threadIdx.x >> 5);
  const int half = lane >> 4;
  const int rl   = lane & 15;
  const int b    = blockIdx.x >> 4;
  const int m0   = (blockIdx.x & 15) * 16;

  v8f acc = {};
  const _Float16* aRow = Aadj + ((size_t)b * 256 + m0 + rl) * 256;
  const _Float16* bRow = Bt + ((size_t)b * NP + t * 16 + rl) * 256;
#pragma unroll
  for (int k0 = 0; k0 < 256; k0 += 32) {
    v16h af = load_fragA(aRow + k0, half);
    v16h bf = load_fragB(bRow + k0, half);
    acc = __builtin_amdgcn_wmma_f32_16x16x32_f16(false, af, false, bf,
                                                 (short)0, acc, false, false);
  }
  const int n = t * 16 + rl;
#pragma unroll
  for (int r = 0; r < 8; ++r) {
    int m = m0 + r + 8 * half;
    outH[((size_t)b * 256 + m) * KP_S + n] = (_Float16)(acc[r] * scale);
  }
}

// ---------------------------------------------------------------------------
// s1[b,l] = go1[b,l,:].(sum_h W1[:,h]) ; s2 similarly with W2
// ---------------------------------------------------------------------------
__global__ void s1s2_k(const float* __restrict__ go1, const float* __restrict__ Wx_w,
                       float* __restrict__ s1, float* __restrict__ s2) {
  int idx = blockIdx.x * blockDim.x + threadIdx.x;
  if (idx >= 8192) return;
  const float* row = go1 + (size_t)idx * NP;
  float a = 0.f, c2 = 0.f;
  for (int c = 0; c < 100; ++c) {
    float w1 = 0.f, w2 = 0.f;
#pragma unroll
    for (int h = 0; h < 5; ++h) {
      w1 += Wx_w[(5 + c) * 5 + h];
      w2 += Wx_w[(105 + c) * 5 + h];
    }
    float g = row[c];
    a += g * w1; c2 += g * w2;
  }
  s1[idx] = a; s2[idx] = c2;
}

// adjS2[b,i,j] = adjW + s1[b,j] + s2[b,i] + sum(Wx_b)
__global__ void adjs2_k(const _Float16* __restrict__ adjWh, const float* __restrict__ s1,
                        const float* __restrict__ s2, const float* __restrict__ Wx_b,
                        _Float16* __restrict__ adjS2h) {
  size_t idx = (size_t)blockIdx.x * blockDim.x + threadIdx.x;
  if (idx >= (size_t)32 * 65536) return;
  int b = (int)(idx >> 16);
  int ij = (int)(idx & 65535);
  int i = ij >> 8, j = ij & 255;
  float c0 = Wx_b[0] + Wx_b[1] + Wx_b[2] + Wx_b[3] + Wx_b[4];
  adjS2h[idx] = (_Float16)((float)adjWh[idx] + s1[b * 256 + j] + s2[b * 256 + i] + c0);
}

// ---------------------------------------------------------------------------
// Final: out1 = (relu(go2)*amask).sum(L)/wn ; out = out1 @ clf_w + clf_b
// ---------------------------------------------------------------------------
__global__ void __launch_bounds__(128) final_k(
    const float* __restrict__ go2, const float* __restrict__ amask,
    const float* __restrict__ clf_w, const float* __restrict__ clf_b,
    float* __restrict__ out) {
  __shared__ float h1[100];
  __shared__ float red[128];
  int b = blockIdx.x, t = threadIdx.x;
  float wn = 0.f;
  for (int l = t; l < 256; l += 128) wn += amask[b * 256 + l];
  red[t] = wn; __syncthreads();
  for (int off = 64; off > 0; off >>= 1) {
    if (t < off) red[t] += red[t + off];
    __syncthreads();
  }
  wn = red[0];
  if (t < 100) {
    float s = 0.f;
    for (int l = 0; l < 256; ++l) {
      float v = fmaxf(go2[((size_t)b * 256 + l) * NP + t], 0.0f);
      s += v * amask[b * 256 + l];
    }
    h1[t] = s / wn;
  }
  __syncthreads();
  if (t < 3) {
    float s = 0.f;
    for (int c = 0; c < 100; ++c) s += h1[c] * clf_w[c * 3 + t];
    out[b * 3 + t] = s + clf_b[t];
  }
}

// ---------------------------------------------------------------------------
// Host launcher
// ---------------------------------------------------------------------------
extern "C" void kernel_launch(void* const* d_in, const int* in_sizes, int n_in,
                              void* d_out, int out_size, void* d_ws, size_t ws_size,
                              hipStream_t stream) {
  (void)in_sizes; (void)n_in; (void)out_size; (void)ws_size;
  const float* seq_out   = (const float*)d_in[0];
  const int*   src_mask  = (const int*)  d_in[1];
  const float* amask     = (const float*)d_in[2];
  const float* short_m   = (const float*)d_in[3];
  const float* ln_a      = (const float*)d_in[4];
  const float* ln_b      = (const float*)d_in[5];
  const float* Wxx_w     = (const float*)d_in[6];
  const float* Wxx_b     = (const float*)d_in[7];
  const float* q_w       = (const float*)d_in[8];
  const float* q_b       = (const float*)d_in[9];
  const float* k_w       = (const float*)d_in[10];
  const float* k_b       = (const float*)d_in[11];
  const float* dense_w   = (const float*)d_in[12];
  const float* dense_b   = (const float*)d_in[13];
  const float* bias_m    = (const float*)d_in[14];
  const float* W_w       = (const float*)d_in[15];
  const float* W_b       = (const float*)d_in[16];
  const float* Wx_w      = (const float*)d_in[17];
  const float* Wx_b      = (const float*)d_in[18];
  const float* clf_w     = (const float*)d_in[19];
  const float* clf_b     = (const float*)d_in[20];
  float* out = (float*)d_out;

  char* ws = (char*)d_ws;
  size_t off = 0;
  auto alloc = [&](size_t bytes) -> void* {
    void* p = ws + off;
    off += (bytes + 255) & ~(size_t)255;
    return p;
  };

  _Float16* xh    = (_Float16*)alloc((size_t)8192 * 768 * 2);
  _Float16* WxxT  = (_Float16*)alloc((size_t)NP * 768 * 2);
  _Float16* qwT   = (_Float16*)alloc((size_t)NP * KP_S * 2);
  _Float16* kwT   = (_Float16*)alloc((size_t)NP * KP_S * 2);
  _Float16* WwT   = (_Float16*)alloc((size_t)NP * KP_S * 2);
  float* bWxx     = (float*)alloc(NP * 4);
  float* bq       = (float*)alloc(NP * 4);
  float* bk       = (float*)alloc(NP * 4);
  float* bW       = (float*)alloc(NP * 4);
  float* gF       = (float*)alloc((size_t)8192 * NP * 4);
  _Float16* gT    = (_Float16*)alloc((size_t)32 * NP * 256 * 2);
  float* go1F     = (float*)alloc((size_t)8192 * NP * 4);
  _Float16* go1T  = (_Float16*)alloc((size_t)32 * NP * 256 * 2);
  float* go2F     = (float*)alloc((size_t)8192 * NP * 4);
  float* aspG     = (float*)alloc((size_t)32 * 20 * 4);
  float* adjF     = (float*)alloc((size_t)32 * 5 * 65536 * 4);
  _Float16* adjSh = (_Float16*)alloc((size_t)32 * 65536 * 2);
  _Float16* adjWh = (_Float16*)alloc((size_t)32 * 65536 * 2);
  _Float16* adjS2h= (_Float16*)alloc((size_t)32 * 65536 * 2);
  float* s1       = (float*)alloc(8192 * 4);
  float* s2       = (float*)alloc(8192 * 4);
  size_t zstart   = off;
  _Float16* gh    = (_Float16*)alloc((size_t)8192 * KP_S * 2);
  _Float16* qh    = (_Float16*)alloc((size_t)32 * 5 * 256 * 32 * 2);
  _Float16* kh    = (_Float16*)alloc((size_t)32 * 5 * 256 * 32 * 2);
  _Float16* Axh   = (_Float16*)alloc((size_t)8192 * KP_S * 2);
  _Float16* Ax2h  = (_Float16*)alloc((size_t)8192 * KP_S * 2);
  size_t zcount   = (off - zstart) / 2;
  _Float16* zbase = (_Float16*)(ws + zstart);

  // 1) zero f16 staging pads (deterministic each launch)
  zero_h_k<<<(int)((zcount + 255) / 256), 256, 0, stream>>>(zbase, zcount);
  // 2) layernorm -> f16
  layernorm_k<<<1024, 256, 0, stream>>>(seq_out, ln_a, ln_b, xh);
  // 3) weight packs
  packWT_k<<<(NP * 768 + 255) / 256, 256, 0, stream>>>(Wxx_w, WxxT, 768, 100, 768);
  packWT_k<<<(NP * KP_S + 255) / 256, 256, 0, stream>>>(q_w, qwT, 100, 100, KP_S);
  packWT_k<<<(NP * KP_S + 255) / 256, 256, 0, stream>>>(k_w, kwT, 100, 100, KP_S);
  packWT_k<<<(NP * KP_S + 255) / 256, 256, 0, stream>>>(W_w, WwT, 100, 100, KP_S);
  packBias_k<<<1, 128, 0, stream>>>(Wxx_b, bWxx, 100);
  packBias_k<<<1, 128, 0, stream>>>(q_b, bq, 100);
  packBias_k<<<1, 128, 0, stream>>>(k_b, bk, 100);
  packBias_k<<<1, 128, 0, stream>>>(W_b, bW, 100);
  // 4) g = LN(x) @ Wxx + b   (f32, f16 row-major, per-batch transposed)
  dense_gemm_k<<<512, 224, 0, stream>>>(xh, 768, WxxT, bWxx, 0, gF, gh, gT, nullptr);
  // 5) aspect -> asp
  aspect_k<<<32, 128, 0, stream>>>(gF, amask, dense_w, dense_b, aspG);
  // 6) q, k per-head padded f16
  dense_gemm_k<<<512, 224, 0, stream>>>(gh, KP_S, qwT, bq, 0, nullptr, nullptr, nullptr, qh);
  dense_gemm_k<<<512, 224, 0, stream>>>(gh, KP_S, kwT, bk, 0, nullptr, nullptr, nullptr, kh);
  // 7) attention + softmax -> adj (B,H,L,L) f32
  attention_k<<<2560, 128, 0, stream>>>(qh, kh, aspG, bias_m, src_mask, short_m, adjF);
  // 8) head reductions
  head_reduce_k<<<8192, 256, 0, stream>>>(adjF, Wx_w, adjSh, adjWh);
  // 9) Ax = adjS @ g / H
  bgemm_k<<<512, 224, 0, stream>>>(adjSh, gT, 0.2f, Axh);
  // 10) go1 = relu(Ax @ W + b)
  dense_gemm_k<<<512, 224, 0, stream>>>(Axh, KP_S, WwT, bW, 1, go1F, nullptr, go1T, nullptr);
  // 11) edge head-sums (rank-1 corrections)
  s1s2_k<<<64, 128, 0, stream>>>(go1F, Wx_w, s1, s2);
  adjs2_k<<<8192, 256, 0, stream>>>(adjWh, s1, s2, Wx_b, adjS2h);
  // 12) Ax2 = adjS2 @ go1 / H ; go2 = relu(Ax2 @ W + b)
  bgemm_k<<<512, 224, 0, stream>>>(adjS2h, go1T, 0.2f, Ax2h);
  dense_gemm_k<<<512, 224, 0, stream>>>(Ax2h, KP_S, WwT, bW, 1, go2F, nullptr, nullptr, nullptr);
  // 13) masked pool + classifier
  final_k<<<32, 128, 0, stream>>>(go2F, amask, clf_w, clf_b, out);
}
